// Speller_19112604467518
// MI455X (gfx1250) — matrix operand order repaired and checked by reading, not compiled, not run on hardware
//
#include <hip/hip_runtime.h>
#include <hip/hip_bf16.h>

// ---------------- problem constants (from reference setup_inputs) ----------
#define B_    64
#define S_    2048
#define ENC_  512
#define H_    512
#define E_    256
#define CTX_  256
#define QK_   128
#define V_    10000
#define NCHUNK_ 8
#define SCHUNK_ (S_ / NCHUNK_)   // 256

typedef __attribute__((ext_vector_type(16))) __bf16 v16bf;
typedef __attribute__((ext_vector_type(8)))  __bf16 v8bf;
typedef __attribute__((ext_vector_type(8)))  float  v8f;

// ---------------------------------------------------------------------------
// WMMA tile loaders per CDNA5 ISA 7.12.2 (wave32).
// A-matrix 16x32 bf16 (M x K), row-major A[m*lda + k]:
//   lanes 0-15  : M=lane,    K in {0..7} U {16..23}
//   lanes 16-31 : M=lane-16, K in {8..15} U {24..31}
// B-matrix 32x16 bf16 (K x N) fed from weight W[N,K] row-major (computes X@W^T):
//   lanes 0-15  : N=lane,    K in {0..7} U {16..23}
//   lanes 16-31 : N=lane-16, K in {8..15} U {24..31}
// C/D 16x16 f32: VGPR r -> M = r + (lane>=16 ? 8 : 0), N = lane&15.
// ---------------------------------------------------------------------------
__device__ __forceinline__ v16bf load_tileA_bf(const __bf16* A, int lda, int mbase, int kbase) {
    int l  = threadIdx.x & 31;
    int m  = mbase + (l & 15);
    int k0 = kbase + ((l >> 4) << 3);          // +8 for upper half-wave
    const __bf16* p = A + (size_t)m * lda + k0;
    v8bf lo = *(const v8bf*)(p);
    v8bf hi = *(const v8bf*)(p + 16);
    v16bf r;
#pragma unroll
    for (int i = 0; i < 8; ++i) { r[i] = lo[i]; r[i + 8] = hi[i]; }
    return r;
}

__device__ __forceinline__ v16bf load_tileA_f32(const float* A, int lda, int mbase, int kbase) {
    int l  = threadIdx.x & 31;
    int m  = mbase + (l & 15);
    int k0 = kbase + ((l >> 4) << 3);
    const float* p = A + (size_t)m * lda + k0;
    v8f lo = *(const v8f*)(p);
    v8f hi = *(const v8f*)(p + 16);
    v16bf r;
#pragma unroll
    for (int i = 0; i < 8; ++i) { r[i] = (__bf16)lo[i]; r[i + 8] = (__bf16)hi[i]; }
    return r;
}

__device__ __forceinline__ v16bf load_tileB_bf(const __bf16* W, int ldb, int nbase, int kbase) {
    int l = threadIdx.x & 31;
    int n = nbase + (l & 15);
    int k0 = kbase + ((l >> 4) << 3);          // +8 for upper half-wave
    const __bf16* p = W + (size_t)n * ldb + k0;
    v8bf lo = *(const v8bf*)(p);
    v8bf hi = *(const v8bf*)(p + 16);
    v16bf r;
#pragma unroll
    for (int i = 0; i < 8; ++i) { r[i] = lo[i]; r[i + 8] = hi[i]; }
    return r;
}

// ---------------------------------------------------------------------------
// bf16 WMMA GEMM, register-blocked over M: one wave computes a 64x16 tile
// (4 accumulators), so each B (weight) tile load feeds 4 WMMAs.
// C[M,N] = act( A1@B1^T (+ A2@B2^T) + bias ).  grid = (N/16, M/64), block = 32.
// ---------------------------------------------------------------------------
__global__ void wmma_gemm_bf16(const __bf16* __restrict__ A1, int lda1,
                               const __bf16* __restrict__ B1, int ldb1, int K1,
                               const __bf16* __restrict__ A2, int lda2,
                               const __bf16* __restrict__ B2, int ldb2, int K2,
                               const float* __restrict__ bias, int act, float slope,
                               float* __restrict__ Cf, long long ldcf,
                               __bf16* __restrict__ Cb, int ldcb) {
    int nb = blockIdx.x * 16;
    int mb = blockIdx.y * 64;
    v8f acc[4] = {};
    for (int k = 0; k < K1; k += 32) {
        __builtin_prefetch(B1 + (size_t)nb * ldb1 + k + 64, 0, 1);
        v16bf b = load_tileB_bf(B1, ldb1, nb, k);
#pragma unroll
        for (int mt = 0; mt < 4; ++mt) {
            v16bf a = load_tileA_bf(A1, lda1, mb + mt * 16, k);
            acc[mt] = __builtin_amdgcn_wmma_f32_16x16x32_bf16(false, a, false, b,
                                                              (short)0, acc[mt], false, false);
        }
    }
    if (A2) {
        for (int k = 0; k < K2; k += 32) {
            __builtin_prefetch(B2 + (size_t)nb * ldb2 + k + 64, 0, 1);
            v16bf b = load_tileB_bf(B2, ldb2, nb, k);
#pragma unroll
            for (int mt = 0; mt < 4; ++mt) {
                v16bf a = load_tileA_bf(A2, lda2, mb + mt * 16, k);
                acc[mt] = __builtin_amdgcn_wmma_f32_16x16x32_bf16(false, a, false, b,
                                                                  (short)0, acc[mt], false, false);
            }
        }
    }
    int l  = threadIdx.x & 31;
    int n  = nb + (l & 15);
    int mo = (l >> 4) * 8;
    float bv = bias ? bias[n] : 0.0f;
#pragma unroll
    for (int mt = 0; mt < 4; ++mt) {
#pragma unroll
        for (int r = 0; r < 8; ++r) {
            int m = mb + mt * 16 + mo + r;
            float v = acc[mt][r] + bv;
            if (act) v = (v >= 0.0f) ? v : slope * v;
            if (Cf) Cf[(size_t)m * ldcf + n] = v;
            if (Cb) Cb[(size_t)m * ldcb + n] = (__bf16)v;
        }
    }
}

// f32-A variant (hoisted key/val encoder projections), same 64x16 blocking.
__global__ void wmma_gemm_f32A(const float* __restrict__ A, int lda,
                               const __bf16* __restrict__ Bm, int ldb, int K,
                               const float* __restrict__ bias, float slope,
                               __bf16* __restrict__ Cb, int ldcb) {
    int nb = blockIdx.x * 16;
    int mb = blockIdx.y * 64;
    v8f acc[4] = {};
    for (int k = 0; k < K; k += 32) {
        __builtin_prefetch(A + (size_t)(mb + 64) * lda + k, 0, 1);
        v16bf b = load_tileB_bf(Bm, ldb, nb, k);
#pragma unroll
        for (int mt = 0; mt < 4; ++mt) {
            v16bf a = load_tileA_f32(A, lda, mb + mt * 16, k);
            acc[mt] = __builtin_amdgcn_wmma_f32_16x16x32_bf16(false, a, false, b,
                                                              (short)0, acc[mt], false, false);
        }
    }
    int l  = threadIdx.x & 31;
    int n  = nb + (l & 15);
    int mo = (l >> 4) * 8;
    float bv = bias ? bias[n] : 0.0f;
#pragma unroll
    for (int mt = 0; mt < 4; ++mt) {
#pragma unroll
        for (int r = 0; r < 8; ++r) {
            int m = mb + mt * 16 + mo + r;
            float v = acc[mt][r] + bv;
            v = (v >= 0.0f) ? v : slope * v;            // leaky 0.3
            Cb[(size_t)m * ldcb + n] = (__bf16)v;
        }
    }
}

// ---------------------------------------------------------------------------
// Small support kernels
// ---------------------------------------------------------------------------
__global__ void f2bf_kernel(const float* __restrict__ src, __bf16* __restrict__ dst, int n) {
    int i = blockIdx.x * blockDim.x + threadIdx.x;
    if (i < n) dst[i] = (__bf16)src[i];
}

__global__ void init_state_kernel(const float* __restrict__ ih0, const float* __restrict__ ic0,
                                  const float* __restrict__ ih1, const float* __restrict__ ic1,
                                  __bf16* __restrict__ h0_bf, float* __restrict__ c0,
                                  __bf16* __restrict__ feat_bf, float* __restrict__ c1,
                                  int* __restrict__ yt) {
    int idx = blockIdx.x * blockDim.x + threadIdx.x;   // B_*H_
    int b = idx >> 9, j = idx & (H_ - 1);
    h0_bf[idx] = (__bf16)ih0[j];
    c0[idx]    = ic0[j];
    feat_bf[(size_t)b * (H_ + CTX_) + j] = (__bf16)ih1[j];   // h1 lives in feat[:, :H]
    c1[idx]    = ic1[j];
    if (idx < B_) yt[idx] = 0;
}

__global__ void embed_gather_kernel(const float* __restrict__ emb, const int* __restrict__ yt,
                                    __bf16* __restrict__ x_bf) {
    int b = blockIdx.x, e = threadIdx.x;               // 256 threads == E_
    x_bf[(size_t)b * (E_ + CTX_) + e] = (__bf16)emb[(size_t)yt[b] * E_ + e];
}

// energy[b,s] = q[b] . key[b,s]; masked with -1e30 beyond state_len
__global__ void energy_kernel(const __bf16* __restrict__ q, const __bf16* __restrict__ key,
                              const int* __restrict__ state_len, float* __restrict__ energy) {
    __shared__ float qs[QK_];
    int b = blockIdx.x, tid = threadIdx.x;
    if (tid < QK_) qs[tid] = (float)q[(size_t)b * QK_ + tid];
    __syncthreads();
    int s = blockIdx.y * blockDim.x + tid;
    const __bf16* kp = key + ((size_t)b * S_ + s) * QK_;
    float sum = 0.0f;
#pragma unroll
    for (int k = 0; k < QK_; k += 8) {
        v8bf kv = *(const v8bf*)(kp + k);
#pragma unroll
        for (int j = 0; j < 8; ++j) sum += qs[k + j] * (float)kv[j];
    }
    if (s >= state_len[b]) sum = -1e30f;
    energy[(size_t)b * S_ + s] = sum;
}

// per-batch softmax stats: stats[b] = {max, 1/sum(exp(e-max))}
__global__ void softmax_stats_kernel(const float* __restrict__ energy, float* __restrict__ stats) {
    __shared__ float red[256];
    int b = blockIdx.x, tid = threadIdx.x;
    const float* e = energy + (size_t)b * S_;
    float mx = -1e30f;
    for (int s = tid; s < S_; s += 256) mx = fmaxf(mx, e[s]);
    red[tid] = mx; __syncthreads();
    for (int w = 128; w > 0; w >>= 1) { if (tid < w) red[tid] = fmaxf(red[tid], red[tid + w]); __syncthreads(); }
    mx = red[0]; __syncthreads();
    float sm = 0.0f;
    for (int s = tid; s < S_; s += 256) sm += __expf(e[s] - mx);
    red[tid] = sm; __syncthreads();
    for (int w = 128; w > 0; w >>= 1) { if (tid < w) red[tid] += red[tid + w]; __syncthreads(); }
    if (tid == 0) { stats[b * 2] = mx; stats[b * 2 + 1] = 1.0f / red[0]; }
}

// alpha for one S-chunk + partial context over that chunk.
// grid = (B, NCHUNK), block = 256 (== SCHUNK == CTX)
__global__ void alpha_ctx_partial_kernel(const float* __restrict__ energy,
                                         const float* __restrict__ stats,
                                         const __bf16* __restrict__ val,
                                         float* __restrict__ alpha_out, int t, int T,
                                         float* __restrict__ part) {
    __shared__ float a_sh[SCHUNK_];
    int b = blockIdx.x, ch = blockIdx.y, tid = threadIdx.x;
    int s0 = ch * SCHUNK_;
    float mx = stats[b * 2], inv = stats[b * 2 + 1];
    int s = s0 + tid;
    float a = __expf(energy[(size_t)b * S_ + s] - mx) * inv;   // masked entries -> 0
    a_sh[tid] = a;
    alpha_out[((size_t)b * T + t) * S_ + s] = a;
    __syncthreads();
    // tid == context channel c; rows are coalesced 256-wide reads
    float acc = 0.0f;
    const __bf16* vp = val + ((size_t)b * S_ + s0) * CTX_ + tid;
    for (int i = 0; i < SCHUNK_; ++i) acc += a_sh[i] * (float)vp[(size_t)i * CTX_];
    part[((size_t)b * NCHUNK_ + ch) * CTX_ + tid] = acc;
}

__global__ void ctx_reduce_kernel(const float* __restrict__ part,
                                  __bf16* __restrict__ x_bf, __bf16* __restrict__ feat_bf) {
    int b = blockIdx.x, c = threadIdx.x;               // 256 threads == CTX_
    float acc = 0.0f;
#pragma unroll
    for (int ch = 0; ch < NCHUNK_; ++ch) acc += part[((size_t)b * NCHUNK_ + ch) * CTX_ + c];
    x_bf[(size_t)b * (E_ + CTX_) + E_ + c]    = (__bf16)acc;
    feat_bf[(size_t)b * (H_ + CTX_) + H_ + c] = (__bf16)acc;
}

// LSTM gate nonlinearity: g pre-activation [B,4H] (bias b_ih already in), add b_hh here
__global__ void lstm_gates_kernel(const float* __restrict__ g, const float* __restrict__ b_hh,
                                  float* __restrict__ c, __bf16* __restrict__ h_out, int h_ld) {
    int idx = blockIdx.x * blockDim.x + threadIdx.x;   // B_*H_
    int b = idx >> 9, j = idx & (H_ - 1);
    const float* gr = g + (size_t)b * 4 * H_;
    float gi = gr[j]            + b_hh[j];
    float gf = gr[H_ + j]       + b_hh[H_ + j];
    float gg = gr[2 * H_ + j]   + b_hh[2 * H_ + j];
    float go = gr[3 * H_ + j]   + b_hh[3 * H_ + j];
    float si = 1.0f / (1.0f + __expf(-gi));
    float sf = 1.0f / (1.0f + __expf(-gf));
    float so = 1.0f / (1.0f + __expf(-go));
    float cn = sf * c[idx] + si * tanhf(gg);
    c[idx] = cn;
    h_out[(size_t)b * h_ld + j] = (__bf16)(so * tanhf(cn));
}

// greedy argmax over V (first index wins ties, like jnp.argmax)
__global__ void argmax_kernel(const float* __restrict__ logits, long long ldl, int* __restrict__ yt) {
    __shared__ float bv[256];
    __shared__ int   bi[256];
    int b = blockIdx.x, tid = threadIdx.x;
    const float* p = logits + (size_t)b * ldl;
    float best = -3.4e38f; int bidx = 0;
    for (int n = tid; n < V_; n += 256) { float v = p[n]; if (v > best) { best = v; bidx = n; } }
    bv[tid] = best; bi[tid] = bidx; __syncthreads();
    for (int w = 128; w > 0; w >>= 1) {
        if (tid < w) {
            if (bv[tid + w] > bv[tid] || (bv[tid + w] == bv[tid] && bi[tid + w] < bi[tid])) {
                bv[tid] = bv[tid + w]; bi[tid] = bi[tid + w];
            }
        }
        __syncthreads();
    }
    if (tid == 0) yt[b] = bi[0];
}

// ---------------------------------------------------------------------------
extern "C" void kernel_launch(void* const* d_in, const int* in_sizes, int n_in,
                              void* d_out, int out_size, void* d_ws, size_t ws_size,
                              hipStream_t stream) {
    const float* hs           = (const float*)d_in[0];
    const float* embedding    = (const float*)d_in[1];
    const float* classifier_b = (const float*)d_in[2];
    const float* Wq    = (const float*)d_in[3];
    const float* bq    = (const float*)d_in[4];
    const float* Wk    = (const float*)d_in[5];
    const float* bk    = (const float*)d_in[6];
    const float* Wc    = (const float*)d_in[7];
    const float* bc    = (const float*)d_in[8];
    const float* Wchar = (const float*)d_in[9];
    const float* bchar = (const float*)d_in[10];
    const float* W_ih0 = (const float*)d_in[11];
    const float* W_hh0 = (const float*)d_in[12];
    const float* b_ih0 = (const float*)d_in[13];
    const float* b_hh0 = (const float*)d_in[14];
    const float* W_ih1 = (const float*)d_in[15];
    const float* W_hh1 = (const float*)d_in[16];
    const float* b_ih1 = (const float*)d_in[17];
    const float* b_hh1 = (const float*)d_in[18];
    const float* init_h0 = (const float*)d_in[19];
    const float* init_c0 = (const float*)d_in[20];
    const float* init_h1 = (const float*)d_in[21];
    const float* init_c1 = (const float*)d_in[22];
    const int*   state_len = (const int*)d_in[23];
    (void)in_sizes; (void)n_in; (void)ws_size;

    const int T = out_size / (B_ * (V_ + S_));        // derive max_iters on host
    float* out_logits = (float*)d_out;                       // [B, T, V]
    float* out_alpha  = out_logits + (size_t)B_ * T * V_;    // [B, T, S]

    // -------- workspace layout --------
    char* ws = (char*)d_ws;
    size_t off = 0;
    auto alloc = [&](size_t bytes) -> void* {
        void* p = ws + off;
        off = (off + bytes + 255) & ~(size_t)255;
        return p;
    };
    __bf16* wk_bf    = (__bf16*)alloc((size_t)QK_ * ENC_ * 2);
    __bf16* wc_bf    = (__bf16*)alloc((size_t)CTX_ * ENC_ * 2);
    __bf16* wq_bf    = (__bf16*)alloc((size_t)QK_ * H_ * 2);
    __bf16* wchar_bf = (__bf16*)alloc((size_t)E_ * (H_ + CTX_) * 2);
    __bf16* wih0_bf  = (__bf16*)alloc((size_t)4 * H_ * (E_ + CTX_) * 2);
    __bf16* whh0_bf  = (__bf16*)alloc((size_t)4 * H_ * H_ * 2);
    __bf16* wih1_bf  = (__bf16*)alloc((size_t)4 * H_ * H_ * 2);
    __bf16* whh1_bf  = (__bf16*)alloc((size_t)4 * H_ * H_ * 2);
    __bf16* emb_bf   = (__bf16*)alloc((size_t)V_ * E_ * 2);
    __bf16* key_bf   = (__bf16*)alloc((size_t)B_ * S_ * QK_ * 2);
    __bf16* val_bf   = (__bf16*)alloc((size_t)B_ * S_ * CTX_ * 2);
    __bf16* x_bf     = (__bf16*)alloc((size_t)B_ * (E_ + CTX_) * 2);   // [emb | context]
    __bf16* feat_bf  = (__bf16*)alloc((size_t)B_ * (H_ + CTX_) * 2);   // [h1 | context]
    __bf16* h0_bf    = (__bf16*)alloc((size_t)B_ * H_ * 2);
    __bf16* hid_bf   = (__bf16*)alloc((size_t)B_ * E_ * 2);
    __bf16* q_bf     = (__bf16*)alloc((size_t)B_ * QK_ * 2);
    float*  c0       = (float*)alloc((size_t)B_ * H_ * 4);
    float*  c1       = (float*)alloc((size_t)B_ * H_ * 4);
    float*  g0       = (float*)alloc((size_t)B_ * 4 * H_ * 4);
    float*  g1       = (float*)alloc((size_t)B_ * 4 * H_ * 4);
    float*  energy   = (float*)alloc((size_t)B_ * S_ * 4);
    float*  stats    = (float*)alloc((size_t)B_ * 2 * 4);
    float*  ctxpart  = (float*)alloc((size_t)B_ * NCHUNK_ * CTX_ * 4);
    int*    yt       = (int*)alloc((size_t)B_ * 4);

    auto cvt = [&](const float* src, __bf16* dst, int n) {
        f2bf_kernel<<<(n + 255) / 256, 256, 0, stream>>>(src, dst, n);
    };
    // -------- one-time: weight conversion to bf16 --------
    cvt(Wk,    wk_bf,    QK_ * ENC_);
    cvt(Wc,    wc_bf,    CTX_ * ENC_);
    cvt(Wq,    wq_bf,    QK_ * H_);
    cvt(Wchar, wchar_bf, E_ * (H_ + CTX_));
    cvt(W_ih0, wih0_bf,  4 * H_ * (E_ + CTX_));
    cvt(W_hh0, whh0_bf,  4 * H_ * H_);
    cvt(W_ih1, wih1_bf,  4 * H_ * H_);
    cvt(W_hh1, whh1_bf,  4 * H_ * H_);
    cvt(embedding, emb_bf, V_ * E_);

    init_state_kernel<<<(B_ * H_) / 256, 256, 0, stream>>>(
        init_h0, init_c0, init_h1, init_c1, h0_bf, c0, feat_bf, c1, yt);

    // -------- hoisted encoder projections: key_s, val_s (bf16, leaky 0.3) --------
    {
        dim3 gk(QK_ / 16, (B_ * S_) / 64);
        wmma_gemm_f32A<<<gk, 32, 0, stream>>>(hs, ENC_, wk_bf, ENC_, ENC_, bk, 0.3f, key_bf, QK_);
        dim3 gv(CTX_ / 16, (B_ * S_) / 64);
        wmma_gemm_f32A<<<gv, 32, 0, stream>>>(hs, ENC_, wc_bf, ENC_, ENC_, bc, 0.3f, val_bf, CTX_);
    }

    // -------- sequential decode --------
    const long long ldlog = (long long)T * V_;
    for (int t = 0; t < T; ++t) {
        // emb = embedding[yt] -> x[:, :E]
        embed_gather_kernel<<<B_, E_, 0, stream>>>(embedding, yt, x_bf);

        // q = leaky(h1 @ Wq^T + bq, 0.3)   (h1 read from feat[:, :H], lda = H+CTX)
        wmma_gemm_bf16<<<dim3(QK_ / 16, B_ / 64), 32, 0, stream>>>(
            feat_bf, H_ + CTX_, wq_bf, H_, H_,
            nullptr, 0, nullptr, 0, 0,
            bq, 1, 0.3f, nullptr, 0, q_bf, QK_);

        // attention: energy -> stats -> alpha + partial context -> context
        energy_kernel<<<dim3(B_, S_ / 256), 256, 0, stream>>>(q_bf, key_bf, state_len, energy);
        softmax_stats_kernel<<<B_, 256, 0, stream>>>(energy, stats);
        alpha_ctx_partial_kernel<<<dim3(B_, NCHUNK_), 256, 0, stream>>>(
            energy, stats, val_bf, out_alpha, t, T, ctxpart);
        ctx_reduce_kernel<<<B_, CTX_, 0, stream>>>(ctxpart, x_bf, feat_bf);

        // LSTM cell 0: g0 = x@W_ih0^T + h0@W_hh0^T + b_ih0
        wmma_gemm_bf16<<<dim3(4 * H_ / 16, B_ / 64), 32, 0, stream>>>(
            x_bf, E_ + CTX_, wih0_bf, E_ + CTX_, E_ + CTX_,
            h0_bf, H_, whh0_bf, H_, H_,
            b_ih0, 0, 0.0f, g0, 4 * H_, nullptr, 0);
        lstm_gates_kernel<<<(B_ * H_) / 256, 256, 0, stream>>>(g0, b_hh0, c0, h0_bf, H_);

        // LSTM cell 1: g1 = h0@W_ih1^T + h1@W_hh1^T + b_ih1
        wmma_gemm_bf16<<<dim3(4 * H_ / 16, B_ / 64), 32, 0, stream>>>(
            h0_bf, H_, wih1_bf, H_, H_,
            feat_bf, H_ + CTX_, whh1_bf, H_, H_,
            b_ih1, 0, 0.0f, g1, 4 * H_, nullptr, 0);
        lstm_gates_kernel<<<(B_ * H_) / 256, 256, 0, stream>>>(g1, b_hh1, c1, feat_bf, H_ + CTX_);

        // hid = leaky(feat @ Wchar^T + bchar, 0.2)
        wmma_gemm_bf16<<<dim3(E_ / 16, B_ / 64), 32, 0, stream>>>(
            feat_bf, H_ + CTX_, wchar_bf, H_ + CTX_, H_ + CTX_,
            nullptr, 0, nullptr, 0, 0,
            bchar, 1, 0.2f, nullptr, 0, hid_bf, E_);

        // logits = hid @ embedding^T + classifier_b  -> d_out[b, t, :]
        wmma_gemm_bf16<<<dim3(V_ / 16, B_ / 64), 32, 0, stream>>>(
            hid_bf, E_, emb_bf, E_, E_,
            nullptr, 0, nullptr, 0, 0,
            classifier_b, 0, 0.0f, out_logits + (size_t)t * V_, ldlog, nullptr, 0);

        // greedy feedback
        argmax_kernel<<<B_, 256, 0, stream>>>(out_logits + (size_t)t * V_, ldlog, yt);
    }
}
